// CosSimRouter_learnable_pad_49778670960797
// MI455X (gfx1250) — matrix operand order repaired and checked by reading, not compiled, not run on hardware
//
#include <hip/hip_runtime.h>
#include <hip/hip_bf16.h>
#include <math.h>

// ---------------- problem constants ----------------
#define DV     4096
#define HIDN   8192
#define NHEADS 16
#define DHEAD  256
#define LVN    576     // GRID*GRID
#define LTN    64
#define SMAX   640     // max rows of cat = LV_sel + LT (padded)
#define RMAX   576     // max rows of rem
#define GAMMA_C 0.065f

#ifndef __has_builtin
#define __has_builtin(x) 0
#endif
#if __has_builtin(__builtin_amdgcn_global_load_async_to_lds_b128) && \
    __has_builtin(__builtin_amdgcn_s_wait_asynccnt)
#define HAVE_ASYNC_LDS 1
#else
#define HAVE_ASYNC_LDS 0
#endif

typedef __bf16 bf16;
typedef __attribute__((ext_vector_type(16))) __bf16 v16bf;
typedef __attribute__((ext_vector_type(8)))  float  v8f;

// gcc-style int4 vector: matches the builtin's parameter type exactly
// ('__attribute__((__vector_size__(4*sizeof(int)))) int __device__ *').
typedef int gv4i __attribute__((vector_size(16)));
typedef __attribute__((address_space(1))) gv4i* gv4i_glob;
typedef __attribute__((address_space(3))) gv4i* gv4i_lds;

struct __attribute__((aligned(16))) U4 { unsigned a, b, c, d; };
struct __attribute__((aligned(16))) F4 { float a, b, c, d; };
union FragBF { v16bf v; unsigned u[8]; };
union Pack16 { bf16 h[16]; U4 q[2]; };

static __device__ __forceinline__ bf16 f2bf(float f) {
  unsigned u = __builtin_bit_cast(unsigned, f);
  unsigned r = u + 0x7FFFu + ((u >> 16) & 1u);      // round-to-nearest-even
  unsigned short h = (unsigned short)(r >> 16);
  return __builtin_bit_cast(bf16, h);
}

// 16-byte global->LDS copy: async DMA (ASYNCcnt) when the toolchain exposes
// the gfx1250 builtin, else VGPR round-trip fallback.
static __device__ __forceinline__ void g2l_copy16(const bf16* g, bf16* l) {
#if HAVE_ASYNC_LDS
  __builtin_amdgcn_global_load_async_to_lds_b128(
      (gv4i_glob)g, (gv4i_lds)l, 0, 0);
#else
  *(U4*)l = *(const U4*)g;
#endif
}
static __device__ __forceinline__ void wait_async_then_barrier() {
#if HAVE_ASYNC_LDS
  __builtin_amdgcn_s_wait_asynccnt(0);
#endif
  __syncthreads();
}

// =====================================================================
// Batched WMMA GEMM:  C = act(alpha * A(bf16,MxK) @ B^T + bias) [+ resid]
// B is either f32 weights (Wf, converted to bf16 in-register while staging
// LDS -> single fp32 HBM read of each weight) or bf16 activations (Wb,
// staged via GLOBAL_LOAD_ASYNC_TO_LDS when available). 128x128x64 tiles,
// 8 wave32 waves per block, double-buffered LDS, v_wmma_f32_16x16x32_bf16.
// =====================================================================
#define TLM 128
#define TLN 128
#define TLK 64
#define LSTR 72   // padded LDS row stride (bf16 units); 144B = 9*16B aligned

__global__ void __launch_bounds__(256)
wmma_gemm_kernel(const bf16* __restrict__ A, long bsA, int lda,
                 const float* __restrict__ Wf, const bf16* __restrict__ Wb,
                 long bsW, int ldw,
                 const float* __restrict__ bias,
                 const float* __restrict__ resid, long bsR, int ldr,
                 float* __restrict__ Cf, bf16* __restrict__ Cb,
                 long bsC, int ldc,
                 int M, int N, int K, float alpha, int act)
{
  __shared__ bf16 lA[2][TLM * LSTR];
  __shared__ bf16 lB[2][TLN * LSTR];

  const int tid = threadIdx.x;
  const int bz  = blockIdx.z;
  const int m0  = blockIdx.y * TLM;
  const int n0  = blockIdx.x * TLN;

  A += (size_t)bz * bsA;
  if (Wf)    Wf    += (size_t)bz * bsW;
  if (Wb)    Wb    += (size_t)bz * bsW;
  if (resid) resid += (size_t)bz * bsR;
  if (Cf)    Cf    += (size_t)bz * bsC;
  if (Cb)    Cb    += (size_t)bz * bsC;

  auto stageA = [&](int k0, int buf) {
#pragma unroll
    for (int i = 0; i < 2; ++i) {
      int c   = tid + i * 256;          // chunk of 16 bf16
      int row = c >> 2;
      int col = (c & 3) << 4;
      bf16* l = &lA[buf][row * LSTR + col];
      if (m0 + row < M) {
        const bf16* g = A + (size_t)(m0 + row) * lda + (k0 + col);
        g2l_copy16(g, l);
        g2l_copy16(g + 8, l + 8);
      } else {
        U4 z{0,0,0,0};
        *(U4*)l = z; *(U4*)(l + 8) = z;
      }
    }
  };
  auto stageB = [&](int k0, int buf) {
#pragma unroll
    for (int i = 0; i < 2; ++i) {
      int c   = tid + i * 256;
      int row = c >> 2;
      int col = (c & 3) << 4;
      bf16* l = &lB[buf][row * LSTR + col];
      if (n0 + row < N) {
        if (Wb) {
          const bf16* g = Wb + (size_t)(n0 + row) * ldw + (k0 + col);
          g2l_copy16(g, l);
          g2l_copy16(g + 8, l + 8);
        } else {
          const float* p = Wf + (size_t)(n0 + row) * ldw + (k0 + col);
          if (k0 + TLK < K) __builtin_prefetch(p + TLK, 0, 1); // global_prefetch_b8
          const F4* pf = (const F4*)p;
          Pack16 pk;
#pragma unroll
          for (int e = 0; e < 4; ++e) {
            F4 f = pf[e];
            pk.h[e * 4 + 0] = f2bf(f.a);
            pk.h[e * 4 + 1] = f2bf(f.b);
            pk.h[e * 4 + 2] = f2bf(f.c);
            pk.h[e * 4 + 3] = f2bf(f.d);
          }
          *(U4*)l = pk.q[0]; *(U4*)(l + 8) = pk.q[1];
        }
      } else {
        U4 z{0,0,0,0};
        *(U4*)l = z; *(U4*)(l + 8) = z;
      }
    }
  };

  const int lane = tid & 31;
  const int wave = tid >> 5;       // 0..7
  const int wm   = wave >> 1;      // 0..3 -> M offset 32*wm
  const int wn   = wave & 1;       // 0..1 -> N offset 64*wn
  const int hlf  = lane >> 4;      // 0/1
  const int l16  = lane & 15;

  v8f acc[2][4] = {};

  stageA(0, 0); stageB(0, 0);
  wait_async_then_barrier();
  const int nk = K / TLK;
  for (int kt = 0; kt < nk; ++kt) {
    const int cur = kt & 1;
    if (kt + 1 < nk) { stageA((kt + 1) * TLK, cur ^ 1); stageB((kt + 1) * TLK, cur ^ 1); }
#pragma unroll
    for (int kk = 0; kk < TLK; kk += 32) {
      FragBF fa[2];
#pragma unroll
      for (int mi = 0; mi < 2; ++mi) {
        const bf16* base = &lA[cur][(wm * 32 + mi * 16 + l16) * LSTR + kk];
#pragma unroll
        for (int v = 0; v < 8; ++v) {
          int kp = ((v < 4) ? 0 : 16) + hlf * 8 + (v & 3) * 2;   // ISA 16-bit A layout
          fa[mi].u[v] = *(const unsigned*)(base + kp);
        }
      }
      FragBF fb[4];
#pragma unroll
      for (int ni = 0; ni < 4; ++ni) {
        const bf16* base = &lB[cur][(wn * 64 + ni * 16 + l16) * LSTR + kk];
#pragma unroll
        for (int v = 0; v < 8; ++v) {
          int kq = ((v < 4) ? 0 : 16) + hlf * 8 + (v & 3) * 2;   // same k order as A
          fb[ni].u[v] = *(const unsigned*)(base + kq);
        }
      }
#pragma unroll
      for (int mi = 0; mi < 2; ++mi)
#pragma unroll
        for (int ni = 0; ni < 4; ++ni)
          acc[mi][ni] = __builtin_amdgcn_wmma_f32_16x16x32_bf16(
              false, fa[mi].v, false, fb[ni].v, (short)0, acc[mi][ni], false, false);
    }
    wait_async_then_barrier();
  }

  const float inv_sqrt2 = 0.7071067811865476f;
#pragma unroll
  for (int mi = 0; mi < 2; ++mi) {
#pragma unroll
    for (int ni = 0; ni < 4; ++ni) {
      int col = n0 + wn * 64 + ni * 16 + l16;
      if (col >= N) continue;
      int rbase = m0 + wm * 32 + mi * 16 + hlf * 8;   // ISA f32 C/D layout
#pragma unroll
      for (int r = 0; r < 8; ++r) {
        int gr = rbase + r;
        if (gr >= M) continue;
        float cv = alpha * acc[mi][ni][r];
        if (bias) cv += bias[col];
        if (act == 1) cv = 0.5f * cv * (1.0f + erff(cv * inv_sqrt2)); // exact GELU
        if (resid) cv += resid[(size_t)gr * ldr + col];
        size_t off = (size_t)gr * ldc + col;
        if (Cf) Cf[off] = cv;
        if (Cb) Cb[off] = f2bf(cv);
      }
    }
  }
}

// =====================================================================
// Small kernels
// =====================================================================
__global__ void __launch_bounds__(256)
rownorm_kernel(const float* __restrict__ X, bf16* __restrict__ Y, int cols)
{
  __shared__ float red[256];
  const int row = blockIdx.x, t = threadIdx.x;
  const float* x = X + (size_t)row * cols;
  float ss = 0.f;
  for (int c = t; c < cols; c += 256) { float v = x[c]; ss += v * v; }
  red[t] = ss; __syncthreads();
  for (int s = 128; s > 0; s >>= 1) { if (t < s) red[t] += red[t + s]; __syncthreads(); }
  float scale = 1.0f / fmaxf(sqrtf(red[0]), 1e-8f);
  bf16* y = Y + (size_t)row * cols;
  for (int c = t; c < cols; c += 256) y[c] = f2bf(x[c] * scale);
}

__global__ void __launch_bounds__(1024)
route_score_kernel(const float* __restrict__ cs, const unsigned char* __restrict__ mask,
                   float* __restrict__ scores)
{
  __shared__ float sh[1024];
  const int t = threadIdx.x;
  float sval = -3.0e38f;
  if (t < LVN) {
    float m = -3.0e38f;
    for (int j = 0; j < LTN; ++j) {
      float v = mask[t * LTN + j] ? cs[t * LTN + j] : 0.0f;
      m = fmaxf(m, v);
    }
    sval = m * 20.0f;   // / TEMP (0.05)
  }
  sh[t] = sval; __syncthreads();
  for (int s = 512; s > 0; s >>= 1) { if (t < s) sh[t] = fmaxf(sh[t], sh[t + s]); __syncthreads(); }
  float mx = sh[0]; __syncthreads();
  float e = (t < LVN) ? __expf(sval - mx) : 0.0f;
  sh[t] = e; __syncthreads();
  for (int s = 512; s > 0; s >>= 1) { if (t < s) sh[t] += sh[t + s]; __syncthreads(); }
  if (t < LVN) scores[t] = e / sh[0];
}

static __device__ __forceinline__ void bitonic_sort_desc_1024(float* k_, int* id_)
{
  const int t = threadIdx.x;
  for (int size = 2; size <= 1024; size <<= 1) {
    for (int stride = size >> 1; stride > 0; stride >>= 1) {
      __syncthreads();
      int j = t ^ stride;
      if (j > t) {
        bool dirDesc = ((t & size) == 0);
        float a = k_[t], b = k_[j];
        bool sw = dirDesc ? (a < b) : (a > b);
        if (sw) { k_[t] = b; k_[j] = a; int ti = id_[t]; id_[t] = id_[j]; id_[j] = ti; }
      }
    }
  }
  __syncthreads();
}

__global__ void __launch_bounds__(1024)
route_select_kernel(const float* __restrict__ scores, int* __restrict__ uniq,
                    int* __restrict__ rem, int* __restrict__ counts)
{
  __shared__ float k_[1024];
  __shared__ int   id_[1024];
  __shared__ unsigned char flag[LVN];
  __shared__ int s_thr;
  const int t = threadIdx.x;
  k_[t] = (t < LVN) ? scores[t] : -3.0e38f;
  id_[t] = t;
  if (t < LVN) flag[t] = 0;
  __syncthreads();
  bitonic_sort_desc_1024(k_, id_);
  if (t == 0) {
    float cum = 0.f; int thr = 0;
    for (int i = 0; i < LVN; ++i) { cum += k_[i]; if (cum <= GAMMA_C) thr++; else break; }
    if (thr < 1) thr = 1;
    s_thr = thr; counts[0] = thr;
  }
  __syncthreads();
  if (t < s_thr) {
    int sel = id_[t];
    int r = sel / 24, c = sel % 24;
    for (int dr = -1; dr <= 1; ++dr)
      for (int dc = -1; dc <= 1; ++dc) {
        if (dr == 0 && dc == 0) continue;
        int rr = min(max(r + dr, 0), 23), cc = min(max(c + dc, 0), 23);
        flag[rr * 24 + cc] = 1;
      }
  }
  __syncthreads();
  if (t == 0) {
    int nu = 0, nr = 0;
    for (int i = 0; i < LVN; ++i) { if (flag[i]) uniq[nu++] = i; else rem[nr++] = i; }
    counts[1] = nu; counts[2] = nr;
  }
}

__global__ void __launch_bounds__(256)
gather_cat_kernel(const float* __restrict__ vf, const float* __restrict__ te,
                  const int* __restrict__ uniq, const int* __restrict__ rem,
                  const int* __restrict__ counts,
                  bf16* __restrict__ catb, float* __restrict__ catf,
                  bf16* __restrict__ remb, float* __restrict__ remf)
{
  const int row = blockIdx.x;
  const int nu = counts[1], nr = counts[2];
  const float* src = nullptr;
  bf16* db; float* df;
  if (row < SMAX) {
    if (row < nu)            src = vf + (size_t)uniq[row] * DV;
    else if (row < nu + LTN) src = te + (size_t)(row - nu) * DV;
    db = catb + (size_t)row * DV; df = catf + (size_t)row * DV;
  } else {
    int r = row - SMAX;
    if (r < nr) src = vf + (size_t)rem[r] * DV;
    db = remb + (size_t)r * DV; df = remf + (size_t)r * DV;
  }
  for (int c = threadIdx.x; c < DV; c += 256) {
    float v = src ? src[c] : 0.0f;
    db[c] = f2bf(v); df[c] = v;
  }
}

__global__ void __launch_bounds__(256)
softmax_att_kernel(const float* __restrict__ S, bf16* __restrict__ P,
                   int sq, int sk, const int* __restrict__ cnt, int addc)
{
  const int q = blockIdx.x, h = blockIdx.z, t = threadIdx.x;
  const float* row = S + ((size_t)h * sq + q) * sk;
  bf16* prow = P + ((size_t)h * sq + q) * sk;
  int klen = cnt[0] + addc; if (klen > sk) klen = sk;
  __shared__ float sh[256];
  float m = -3.0e38f;
  for (int j = t; j < klen; j += 256) m = fmaxf(m, row[j]);
  sh[t] = m; __syncthreads();
  for (int s = 128; s > 0; s >>= 1) { if (t < s) sh[t] = fmaxf(sh[t], sh[t + s]); __syncthreads(); }
  float mx = sh[0]; __syncthreads();
  float sum = 0.f;
  for (int j = t; j < klen; j += 256) sum += __expf(row[j] - mx);
  sh[t] = sum; __syncthreads();
  for (int s = 128; s > 0; s >>= 1) { if (t < s) sh[t] += sh[t + s]; __syncthreads(); }
  float inv = 1.0f / sh[0];
  for (int j = t; j < sk; j += 256)
    prow[j] = f2bf(j < klen ? __expf(row[j] - mx) * inv : 0.0f);
}

__global__ void __launch_bounds__(256)
transpose_v_kernel(const bf16* __restrict__ src, int rowStride, int colOff,
                   bf16* __restrict__ dst, int kdim)
{
  const int idx = blockIdx.x * 256 + threadIdx.x;
  const int total = NHEADS * DHEAD * kdim;
  if (idx >= total) return;
  int k = idx % kdim;
  int he = idx / kdim;   // h*DHEAD + e
  dst[(size_t)he * kdim + k] = src[(size_t)k * rowStride + colOff + he];
}

__global__ void __launch_bounds__(256)
layernorm_kernel(const float* __restrict__ X, const float* __restrict__ g,
                 const float* __restrict__ b, float* __restrict__ Yf,
                 bf16* __restrict__ Yb, int cols)
{
  __shared__ float rs[256], rq[256];
  const int row = blockIdx.x, t = threadIdx.x;
  const float* x = X + (size_t)row * cols;
  float s = 0.f, q = 0.f;
  for (int c = t; c < cols; c += 256) { float v = x[c]; s += v; q += v * v; }
  rs[t] = s; rq[t] = q; __syncthreads();
  for (int st = 128; st > 0; st >>= 1) {
    if (t < st) { rs[t] += rs[t + st]; rq[t] += rq[t + st]; }
    __syncthreads();
  }
  float mean = rs[0] / cols;
  float var  = rq[0] / cols - mean * mean;
  float rstd = rsqrtf(var + 1e-5f);
  for (int c = t; c < cols; c += 256) {
    float v = (x[c] - mean) * rstd * g[c] + b[c];
    if (Yf) Yf[(size_t)row * cols + c] = v;
    if (Yb) Yb[(size_t)row * cols + c] = f2bf(v);
  }
}

__global__ void __launch_bounds__(256)
sp_score_kernel(const float* __restrict__ X, const float* __restrict__ w,
                const float* __restrict__ b, const int* __restrict__ counts,
                float* __restrict__ sc)
{
  __shared__ float red[256];
  const int row = blockIdx.x, t = threadIdx.x;
  float s = 0.f;
  for (int c = t; c < DV; c += 256) s += X[(size_t)row * DV + c] * w[c];
  red[t] = s; __syncthreads();
  for (int st = 128; st > 0; st >>= 1) { if (t < st) red[t] += red[t + st]; __syncthreads(); }
  if (t == 0) {
    float v = red[0] + b[0];
    float sig = 1.0f / (1.0f + __expf(-v));
    sc[row] = (row < counts[2]) ? sig : -3.0e38f;
  }
}

__global__ void __launch_bounds__(1024)
final_select_kernel(const float* __restrict__ sc, const int* __restrict__ uniq,
                    const int* __restrict__ rem, int* __restrict__ counts,
                    int* __restrict__ finalIdx)
{
  __shared__ float k_[1024];
  __shared__ int   id_[1024];
  __shared__ unsigned char flag[LVN];
  const int t = threadIdx.x;
  k_[t] = (t < LVN) ? sc[t] : -3.0e38f;
  id_[t] = t;
  if (t < LVN) flag[t] = 0;
  __syncthreads();
  bitonic_sort_desc_1024(k_, id_);
  const int thr = counts[0];
  const int nu  = counts[1];
  const int nr  = counts[2];
  int ksel = (int)((double)thr * 0.3); if (ksel < 1) ksel = 1;
  if (t < nu) flag[uniq[t]] = 1;
  if (t < ksel && id_[t] < nr) flag[rem[id_[t]]] = 1;
  __syncthreads();
  if (t == 0) {
    int nf = 0;
    for (int i = 0; i < LVN; ++i) if (flag[i]) finalIdx[nf++] = i;
    counts[3] = nf;
  }
}

__global__ void __launch_bounds__(256)
output_gather_kernel(const float* __restrict__ vf, const int* __restrict__ finalIdx,
                     const int* __restrict__ counts, float* __restrict__ out,
                     long outElems)
{
  const int row = blockIdx.x;
  const int nf = counts[3];
  for (int c = threadIdx.x; c < DV; c += 256) {
    long o = (long)row * DV + c;
    if (o < outElems)
      out[o] = (row < nf) ? vf[(size_t)finalIdx[row] * DV + c] : 0.0f;
  }
}

// =====================================================================
// Host-side orchestration
// =====================================================================
static inline void launch_gemm(hipStream_t s,
    const bf16* A, long bsA, int lda,
    const float* Wf, const bf16* Wb, long bsW, int ldw,
    const float* bias, const float* resid, long bsR, int ldr,
    float* Cf, bf16* Cb, long bsC, int ldc,
    int M, int N, int K, float alpha, int act, int batch)
{
  dim3 g((N + TLN - 1) / TLN, (M + TLM - 1) / TLM, batch), b(256);
  wmma_gemm_kernel<<<g, b, 0, s>>>(A, bsA, lda, Wf, Wb, bsW, ldw, bias,
                                   resid, bsR, ldr, Cf, Cb, bsC, ldc,
                                   M, N, K, alpha, act);
}

extern "C" void kernel_launch(void* const* d_in, const int* in_sizes, int n_in,
                              void* d_out, int out_size, void* d_ws, size_t ws_size,
                              hipStream_t stream)
{
  (void)in_sizes; (void)n_in; (void)ws_size;
  const float* vf     = (const float*)d_in[0];
  const float* te     = (const float*)d_in[1];
  const unsigned char* mask = (const unsigned char*)d_in[2];
  const float* sa1_w  = (const float*)d_in[3];
  const float* sa1_b  = (const float*)d_in[4];
  const float* sa1_ow = (const float*)d_in[5];
  const float* sa1_ob = (const float*)d_in[6];
  const float* sa2_w  = (const float*)d_in[7];
  const float* sa2_b  = (const float*)d_in[8];
  const float* sa2_ow = (const float*)d_in[9];
  const float* sa2_ob = (const float*)d_in[10];
  const float* ca_w   = (const float*)d_in[11];
  const float* ca_b   = (const float*)d_in[12];
  const float* ca_ow  = (const float*)d_in[13];
  const float* ca_ob  = (const float*)d_in[14];
  const float* ln1_g  = (const float*)d_in[15];
  const float* ln1_b  = (const float*)d_in[16];
  const float* ln2_g  = (const float*)d_in[17];
  const float* ln2_b  = (const float*)d_in[18];
  const float* ln3_g  = (const float*)d_in[19];
  const float* ln3_b  = (const float*)d_in[20];
  const float* ln4_g  = (const float*)d_in[21];
  const float* ln4_b  = (const float*)d_in[22];
  const float* ffn_w1 = (const float*)d_in[23];
  const float* ffn_b1 = (const float*)d_in[24];
  const float* ffn_w2 = (const float*)d_in[25];
  const float* ffn_b2 = (const float*)d_in[26];
  const float* sp_w   = (const float*)d_in[27];
  const float* sp_b   = (const float*)d_in[28];

  char* wsb = (char*)d_ws;
  size_t off = 0;
  auto alloc = [&](size_t n) -> char* {
    char* p = wsb + off;
    off += (n + 255) & ~(size_t)255;
    return p;
  };

  bf16*  vn_bf  = (bf16*) alloc((size_t)LVN * DV * 2);
  bf16*  tn_bf  = (bf16*) alloc((size_t)LTN * DV * 2);
  float* cs_f   = (float*)alloc((size_t)LVN * LTN * 4);
  float* rsc    = (float*)alloc((size_t)LVN * 4);
  int*   uniq   = (int*)  alloc((size_t)LVN * 4);
  int*   rem    = (int*)  alloc((size_t)LVN * 4);
  int*   counts = (int*)  alloc(16 * 4);
  int*   finIdx = (int*)  alloc(1024 * 4);
  float* spc    = (float*)alloc((size_t)LVN * 4);
  bf16*  cat_bf = (bf16*) alloc((size_t)SMAX * DV * 2);
  float* cat_f  = (float*)alloc((size_t)SMAX * DV * 4);
  bf16*  rem_bf = (bf16*) alloc((size_t)RMAX * DV * 2);
  float* rem_f  = (float*)alloc((size_t)RMAX * DV * 4);
  bf16*  qkv_bf = (bf16*) alloc((size_t)SMAX * 3 * DV * 2);
  float* att_s  = (float*)alloc((size_t)NHEADS * SMAX * SMAX * 4);
  bf16*  att_p  = (bf16*) alloc((size_t)NHEADS * SMAX * SMAX * 2);
  bf16*  vt_bf  = (bf16*) alloc((size_t)NHEADS * DHEAD * SMAX * 2);
  bf16*  ao_bf  = (bf16*) alloc((size_t)SMAX * DV * 2);
  float* proj_f = (float*)alloc((size_t)SMAX * DV * 4);
  float* x_f    = (float*)alloc((size_t)SMAX * DV * 4);
  bf16*  x_bf   = (bf16*) alloc((size_t)SMAX * DV * 2);
  float* y_f    = (float*)alloc((size_t)RMAX * DV * 4);
  bf16*  y_bf   = (bf16*) alloc((size_t)RMAX * DV * 2);
  float* x2_f   = (float*)alloc((size_t)RMAX * DV * 4);
  bf16*  x2_bf  = (bf16*) alloc((size_t)RMAX * DV * 2);
  float* x3_f   = (float*)alloc((size_t)RMAX * DV * 4);
  bf16*  q_bf   = (bf16*) alloc((size_t)RMAX * DV * 2);
  bf16*  kv_bf  = (bf16*) alloc((size_t)SMAX * 2 * DV * 2);
  bf16*  h1_bf  = (bf16*) alloc((size_t)RMAX * HIDN * 2);

  // ---------------- routing ----------------
  rownorm_kernel<<<LVN, 256, 0, stream>>>(vf, vn_bf, DV);
  rownorm_kernel<<<LTN, 256, 0, stream>>>(te, tn_bf, DV);
  launch_gemm(stream, vn_bf, 0, DV, nullptr, tn_bf, 0, DV,
              nullptr, nullptr, 0, 0, cs_f, nullptr, 0, LTN,
              LVN, LTN, DV, 1.0f, 0, 1);
  route_score_kernel <<<1, 1024, 0, stream>>>(cs_f, mask, rsc);
  route_select_kernel<<<1, 1024, 0, stream>>>(rsc, uniq, rem, counts);
  gather_cat_kernel<<<SMAX + RMAX, 256, 0, stream>>>(vf, te, uniq, rem, counts,
                                                     cat_bf, cat_f, rem_bf, rem_f);

  const float ascale = 0.0625f;  // 1/sqrt(DHEAD)

  // ---------------- SA1 on cat (S = SMAX, klen = nu + LT) ----------------
  launch_gemm(stream, cat_bf, 0, DV, sa1_w, nullptr, 0, DV, sa1_b,
              nullptr, 0, 0, nullptr, qkv_bf, 0, 3 * DV,
              SMAX, 3 * DV, DV, 1.0f, 0, 1);
  launch_gemm(stream, qkv_bf, DHEAD, 3 * DV, nullptr, qkv_bf + DV, DHEAD, 3 * DV,
              nullptr, nullptr, 0, 0, att_s, nullptr, (long)SMAX * SMAX, SMAX,
              SMAX, SMAX, DHEAD, ascale, 0, NHEADS);
  softmax_att_kernel<<<dim3(SMAX, 1, NHEADS), 256, 0, stream>>>(att_s, att_p, SMAX, SMAX, counts + 1, LTN);
  transpose_v_kernel<<<(NHEADS * DHEAD * SMAX + 255) / 256, 256, 0, stream>>>(qkv_bf, 3 * DV, 2 * DV, vt_bf, SMAX);
  launch_gemm(stream, att_p, (long)SMAX * SMAX, SMAX, nullptr, vt_bf, (long)DHEAD * SMAX, SMAX,
              nullptr, nullptr, 0, 0, nullptr, ao_bf, DHEAD, DV,
              SMAX, DHEAD, SMAX, 1.0f, 0, NHEADS);
  launch_gemm(stream, ao_bf, 0, DV, sa1_ow, nullptr, 0, DV, sa1_ob,
              cat_f, 0, DV, proj_f, nullptr, 0, DV,
              SMAX, DV, DV, 1.0f, 0, 1);
  layernorm_kernel<<<SMAX, 256, 0, stream>>>(proj_f, ln1_g, ln1_b, x_f, x_bf, DV);

  // ---------------- SA2 on rem (S = RMAX, klen = nr) ----------------
  launch_gemm(stream, rem_bf, 0, DV, sa2_w, nullptr, 0, DV, sa2_b,
              nullptr, 0, 0, nullptr, qkv_bf, 0, 3 * DV,
              RMAX, 3 * DV, DV, 1.0f, 0, 1);
  launch_gemm(stream, qkv_bf, DHEAD, 3 * DV, nullptr, qkv_bf + DV, DHEAD, 3 * DV,
              nullptr, nullptr, 0, 0, att_s, nullptr, (long)RMAX * RMAX, RMAX,
              RMAX, RMAX, DHEAD, ascale, 0, NHEADS);
  softmax_att_kernel<<<dim3(RMAX, 1, NHEADS), 256, 0, stream>>>(att_s, att_p, RMAX, RMAX, counts + 2, 0);
  transpose_v_kernel<<<(NHEADS * DHEAD * RMAX + 255) / 256, 256, 0, stream>>>(qkv_bf, 3 * DV, 2 * DV, vt_bf, RMAX);
  launch_gemm(stream, att_p, (long)RMAX * RMAX, RMAX, nullptr, vt_bf, (long)DHEAD * RMAX, RMAX,
              nullptr, nullptr, 0, 0, nullptr, ao_bf, DHEAD, DV,
              RMAX, DHEAD, RMAX, 1.0f, 0, NHEADS);
  launch_gemm(stream, ao_bf, 0, DV, sa2_ow, nullptr, 0, DV, sa2_ob,
              rem_f, 0, DV, proj_f, nullptr, 0, DV,
              RMAX, DV, DV, 1.0f, 0, 1);
  layernorm_kernel<<<RMAX, 256, 0, stream>>>(proj_f, ln3_g, ln3_b, y_f, y_bf, DV);

  // ---------------- Cross-attention: q from y, kv from x ----------------
  launch_gemm(stream, y_bf, 0, DV, ca_w, nullptr, 0, DV, ca_b,
              nullptr, 0, 0, nullptr, q_bf, 0, DV,
              RMAX, DV, DV, 1.0f, 0, 1);
  launch_gemm(stream, x_bf, 0, DV, ca_w + (size_t)DV * DV, nullptr, 0, DV, ca_b + DV,
              nullptr, 0, 0, nullptr, kv_bf, 0, 2 * DV,
              SMAX, 2 * DV, DV, 1.0f, 0, 1);
  launch_gemm(stream, q_bf, DHEAD, DV, nullptr, kv_bf, DHEAD, 2 * DV,
              nullptr, nullptr, 0, 0, att_s, nullptr, (long)RMAX * SMAX, SMAX,
              RMAX, SMAX, DHEAD, ascale, 0, NHEADS);
  softmax_att_kernel<<<dim3(RMAX, 1, NHEADS), 256, 0, stream>>>(att_s, att_p, RMAX, SMAX, counts + 1, LTN);
  transpose_v_kernel<<<(NHEADS * DHEAD * SMAX + 255) / 256, 256, 0, stream>>>(kv_bf, 2 * DV, DV, vt_bf, SMAX);
  launch_gemm(stream, att_p, (long)RMAX * SMAX, SMAX, nullptr, vt_bf, (long)DHEAD * SMAX, SMAX,
              nullptr, nullptr, 0, 0, nullptr, ao_bf, DHEAD, DV,
              RMAX, DHEAD, SMAX, 1.0f, 0, NHEADS);
  launch_gemm(stream, ao_bf, 0, DV, ca_ow, nullptr, 0, DV, ca_ob,
              y_f, 0, DV, proj_f, nullptr, 0, DV,
              RMAX, DV, DV, 1.0f, 0, 1);
  layernorm_kernel<<<RMAX, 256, 0, stream>>>(proj_f, ln2_g, ln2_b, x2_f, x2_bf, DV);

  // ---------------- FFN (exact GELU fused in epilogue) ----------------
  launch_gemm(stream, x2_bf, 0, DV, ffn_w1, nullptr, 0, DV, ffn_b1,
              nullptr, 0, 0, nullptr, h1_bf, 0, HIDN,
              RMAX, HIDN, DV, 1.0f, 1, 1);
  launch_gemm(stream, h1_bf, 0, HIDN, ffn_w2, nullptr, 0, HIDN, ffn_b2,
              x2_f, 0, DV, proj_f, nullptr, 0, DV,
              RMAX, DV, HIDN, 1.0f, 0, 1);
  layernorm_kernel<<<RMAX, 256, 0, stream>>>(proj_f, ln4_g, ln4_b, x3_f, nullptr, DV);

  // ---------------- scoring, expansion, final gather ----------------
  sp_score_kernel   <<<RMAX, 256, 0, stream>>>(x3_f, sp_w, sp_b, counts, spc);
  final_select_kernel<<<1, 1024, 0, stream>>>(spc, uniq, rem, counts, finIdx);
  int outRows = (out_size + DV - 1) / DV;
  if (outRows < 1) outRows = 1;
  output_gather_kernel<<<outRows, 256, 0, stream>>>(vf, finIdx, counts,
                                                    (float*)d_out, (long)out_size);
}
